// ExtWholeBlock_18554258719387
// MI455X (gfx1250) — matrix operand order repaired and checked by reading, not compiled, not run on hardware
//
#include <hip/hip_runtime.h>
#include <hip/hip_bf16.h>

typedef __bf16 bf16;
typedef __attribute__((ext_vector_type(16))) __bf16 v16bf;
typedef __attribute__((ext_vector_type(8)))  float  v8f;

constexpr int cB = 4, cF = 4096, cC = 512, cN = 1024, cH = 8, cD = 64, cHID = 1024;
constexpr float EPS_LN = 1e-5f;

// gfx1250 async global->LDS copy support (device pass only)
#if defined(__gfx1250__) && __has_builtin(__builtin_amdgcn_global_load_async_to_lds_b128)
#define USE_ASYNC_LDS 1
typedef int int4v __attribute__((ext_vector_type(4)));
typedef __attribute__((address_space(1))) int4v as1_i4;   // global int4
typedef __attribute__((address_space(3))) int4v as3_i4;   // LDS int4
#else
#define USE_ASYNC_LDS 0
#endif

__device__ __forceinline__ void wait_async_then_barrier() {
#if USE_ASYNC_LDS
#if __has_builtin(__builtin_amdgcn_s_wait_asynccnt)
  __builtin_amdgcn_s_wait_asynccnt(0);
#else
  asm volatile("s_wait_asynccnt 0x0" ::: "memory");
#endif
#endif
  __syncthreads();
}

// ---------------------------------------------------------------------------
// LayerNorm over rows of length 512, one block (256 threads) per row, bf16 out
// ---------------------------------------------------------------------------
__global__ __launch_bounds__(256) void ln512_kernel(const float* __restrict__ x,
                                                    const float* __restrict__ w,
                                                    const float* __restrict__ b,
                                                    bf16* __restrict__ out) {
  __shared__ float red[256];
  __shared__ float stat[2];
  const int tid = threadIdx.x;
  const long long row = blockIdx.x;
  const float* xr = x + row * cC;
  float v0 = xr[tid];
  float v1 = xr[tid + 256];
  red[tid] = v0 + v1;
  __syncthreads();
  for (int s = 128; s > 0; s >>= 1) { if (tid < s) red[tid] += red[tid + s]; __syncthreads(); }
  if (tid == 0) stat[0] = red[0];
  __syncthreads();
  red[tid] = v0 * v0 + v1 * v1;
  __syncthreads();
  for (int s = 128; s > 0; s >>= 1) { if (tid < s) red[tid] += red[tid + s]; __syncthreads(); }
  if (tid == 0) stat[1] = red[0];
  __syncthreads();
  const float mean = stat[0] * (1.0f / cC);
  const float var  = stat[1] * (1.0f / cC) - mean * mean;
  const float rstd = rsqrtf(var + EPS_LN);
  bf16* orow = out + row * cC;
  orow[tid]       = (bf16)((v0 - mean) * rstd * w[tid] + b[tid]);
  orow[tid + 256] = (bf16)((v1 - mean) * rstd * w[tid + 256] + b[tid + 256]);
}

// ---------------------------------------------------------------------------
// f32 [K,N] -> bf16 [N,K] transposed weight conversion (32x32 LDS tiles)
// ---------------------------------------------------------------------------
__global__ __launch_bounds__(256) void cvt_bf16_t_kernel(const float* __restrict__ in,
                                                         bf16* __restrict__ out,
                                                         int K, int N) {
  __shared__ float t[32][33];
  const int n0 = blockIdx.x * 32, k0 = blockIdx.y * 32;
  const int tx = threadIdx.x & 31, ty = threadIdx.x >> 5;
  for (int j = ty; j < 32; j += 8) t[j][tx] = in[(long long)(k0 + j) * N + n0 + tx];
  __syncthreads();
  for (int j = ty; j < 32; j += 8) out[(long long)(n0 + j) * K + k0 + tx] = (bf16)t[tx][j];
}

// bf16 [R,Ccols] -> bf16 [Ccols,R]
__global__ __launch_bounds__(256) void transpose_bf16_kernel(const bf16* __restrict__ in,
                                                             bf16* __restrict__ out,
                                                             int R, int Ccols) {
  __shared__ bf16 t[32][33];
  const int c0 = blockIdx.x * 32, r0 = blockIdx.y * 32;
  const int tx = threadIdx.x & 31, ty = threadIdx.x >> 5;
  for (int j = ty; j < 32; j += 8) t[j][tx] = in[(long long)(r0 + j) * Ccols + c0 + tx];
  __syncthreads();
  for (int j = ty; j < 32; j += 8) out[(long long)(c0 + j) * R + r0 + tx] = t[tx][j];
}

// ---------------------------------------------------------------------------
// WMMA GEMM: D = A(bf16 [M,K]) * B^T(bf16 [N,K] row-major) [+bias] + epilogue.
// 256 threads = 8 wave32; block tile BM_ x BN_; each wave owns 32x32 (2x2 frags).
// KC=64, double-buffered LDS, staging via GLOBAL_LOAD_ASYNC_TO_LDS_B128 when
// available (ASYNCcnt + barrier), else batched global->VGPR->LDS copies.
// MODE: 0=f32, 1=bf16, 2=f32+residual, 3=bf16 exact GELU, 4=bf16 row-scaled
// ---------------------------------------------------------------------------
#define KC 64

struct GemmP {
  const bf16* A;  int lda;  long long sA;   // batch stride (elements)
  const bf16* B;  int ldb;  long long sB;   // B is [N,K] row-major
  const float* bias;                        // [N] or nullptr
  float* Of;  bf16* Ob;  int ldc;  long long sC;
  const float* resid;  int ldr;             // residual (f32)
  const float* rowsum; long long sRS;       // per-row L1 sums (MODE 4)
  int M, N, K;
};

template <int BM_, int BN_, int MODE>
__global__ __launch_bounds__(256) void wmma_gemm_kernel(GemmP p) {
  constexpr int WN = BN_ / 32;              // waves across N
  static_assert((BM_ / 32) * WN == 8, "8 waves per block");
  constexpr int LDK = KC + 8;               // row stride (144B -> 16B aligned rows)
  __shared__ __align__(16) bf16 As[2][BM_][LDK];
  __shared__ __align__(16) bf16 Bs[2][BN_][LDK];

  const int batch = blockIdx.z;
  const bf16* A = p.A + (long long)batch * p.sA;
  const bf16* B = p.B + (long long)batch * p.sB;
  const int tileM = blockIdx.y * BM_;
  const int tileN = blockIdx.x * BN_;
  const int tid  = threadIdx.x;
  const int wave = tid >> 5;
  const int lane = tid & 31;
  const int wm = wave / WN;
  const int wn = wave % WN;
  const int l15 = lane & 15;
  const int k0  = (lane < 16) ? 0 : 8;      // unused; A frags load with kb0 as well
  const int kb0 = (lane < 16) ? 0 : 16;     // A/B-frag K phase (contiguous k = 16h + i)

  constexpr int ACH = BM_ * (KC / 8) / 256; // 16B chunks per thread (A)
  constexpr int BCH = BN_ * (KC / 8) / 256; // 16B chunks per thread (B)

  // Stage one K-tile of A and B into LDS buffer `buf` (straight-line).
  auto stage = [&](int buf, int kb) {
#if USE_ASYNC_LDS
#pragma unroll
    for (int i = 0; i < ACH; ++i) {
      const int c  = tid + i * 256;
      const int r  = c / (KC / 8);
      const int kq = (c % (KC / 8)) * 8;
      const bf16* src = A + (long long)(tileM + r) * p.lda + kb + kq;
      __builtin_amdgcn_global_load_async_to_lds_b128(
          (as1_i4*)src, (as3_i4*)&As[buf][r][kq], 0, 0);
      __builtin_prefetch(src + 2 * KC, 0, 1);
    }
#pragma unroll
    for (int i = 0; i < BCH; ++i) {
      const int c  = tid + i * 256;
      const int n  = c / (KC / 8);
      const int kq = (c % (KC / 8)) * 8;
      const bf16* src = B + (long long)(tileN + n) * p.ldb + kb + kq;
      __builtin_amdgcn_global_load_async_to_lds_b128(
          (as1_i4*)src, (as3_i4*)&Bs[buf][n][kq], 0, 0);
      __builtin_prefetch(src + 2 * KC, 0, 1);
    }
#else
    uint4 ta[ACH], tb[BCH];
#pragma unroll
    for (int i = 0; i < ACH; ++i) {
      const int c  = tid + i * 256;
      const int r  = c / (KC / 8);
      const int kq = (c % (KC / 8)) * 8;
      const bf16* src = A + (long long)(tileM + r) * p.lda + kb + kq;
      ta[i] = *(const uint4*)src;
      __builtin_prefetch(src + 2 * KC, 0, 1);
    }
#pragma unroll
    for (int i = 0; i < BCH; ++i) {
      const int c  = tid + i * 256;
      const int n  = c / (KC / 8);
      const int kq = (c % (KC / 8)) * 8;
      const bf16* src = B + (long long)(tileN + n) * p.ldb + kb + kq;
      tb[i] = *(const uint4*)src;
      __builtin_prefetch(src + 2 * KC, 0, 1);
    }
#pragma unroll
    for (int i = 0; i < ACH; ++i) {
      const int c = tid + i * 256;
      *(uint4*)&As[buf][c / (KC / 8)][(c % (KC / 8)) * 8] = ta[i];
    }
#pragma unroll
    for (int i = 0; i < BCH; ++i) {
      const int c = tid + i * 256;
      *(uint4*)&Bs[buf][c / (KC / 8)][(c % (KC / 8)) * 8] = tb[i];
    }
#endif
  };

  v8f acc[2][2] = {};

  stage(0, 0);

  int cur = 0;
  for (int kb = 0; kb < p.K; kb += KC, cur ^= 1) {
    wait_async_then_barrier();
    if (kb + KC < p.K) {   // uniform scalar branch
      stage(cur ^ 1, kb + KC);
    }
#pragma unroll
    for (int kk = 0; kk < KC; kk += 32) {
      v16bf a[2], bv[2];
#pragma unroll
      for (int t = 0; t < 2; ++t) {
        const int ra = wm * 32 + t * 16 + l15;
#pragma unroll
        for (int i = 0; i < 16; ++i) a[t][i] = As[cur][ra][kk + kb0 + i];
        const int nb = wn * 32 + t * 16 + l15;
#pragma unroll
        for (int i = 0; i < 16; ++i) bv[t][i] = Bs[cur][nb][kk + kb0 + i];
      }
#pragma unroll
      for (int tm = 0; tm < 2; ++tm)
#pragma unroll
        for (int tn = 0; tn < 2; ++tn)
          acc[tm][tn] = __builtin_amdgcn_wmma_f32_16x16x32_bf16(
              false, a[tm], false, bv[tn], (short)0, acc[tm][tn], false, false);
    }
  }

  // ---- epilogue (D layout: lane holds col l15; rows (lane>=16?8:0)+r) ----
  const int rb = (lane >> 4) << 3;
#pragma unroll
  for (int tm = 0; tm < 2; ++tm) {
    const int mb = tileM + wm * 32 + tm * 16 + rb;
#pragma unroll
    for (int tn = 0; tn < 2; ++tn) {
      const int col = tileN + wn * 32 + tn * 16 + l15;
      const float biasv = p.bias ? p.bias[col] : 0.0f;
#pragma unroll
      for (int r = 0; r < 8; ++r) {
        const int row = mb + r;
        const float v = acc[tm][tn][r] + biasv;
        const long long idx = (long long)batch * p.sC + (long long)row * p.ldc + col;
        if constexpr (MODE == 0) {
          p.Of[idx] = v;
        } else if constexpr (MODE == 1) {
          p.Ob[idx] = (bf16)v;
        } else if constexpr (MODE == 2) {
          p.Of[idx] = v + p.resid[(long long)row * p.ldr + col];
        } else if constexpr (MODE == 3) {
          p.Ob[idx] = (bf16)(0.5f * v * (1.0f + erff(v * 0.70710678118654752f)));
        } else {
          const float s = 1.0f / (1e-7f + p.rowsum[(long long)batch * p.sRS + row]);
          p.Ob[idx] = (bf16)(v * s);
        }
      }
    }
  }
}

// ---------------------------------------------------------------------------
// Column softmax stats over F (axis=-2): online max & sum(exp) per (h,n)
// logits [H,F,N]; grid (N/64, H), block 256 (64 n-lanes x 4 f-groups)
// ---------------------------------------------------------------------------
__global__ __launch_bounds__(256) void colstats_kernel(const float* __restrict__ logits,
                                                       float* __restrict__ cmax,
                                                       float* __restrict__ csum) {
  const int h  = blockIdx.y;
  const int nl = threadIdx.x & 63;
  const int fg = threadIdx.x >> 6;
  const int n  = blockIdx.x * 64 + nl;
  const float* base = logits + (long long)h * cF * cN + n;
  float m = -3.4e38f, s = 0.0f;
  for (int f = fg; f < cF; f += 4) {
    const float x = base[(long long)f * cN];
    const float mn = fmaxf(m, x);
    s = s * expf(m - mn) + expf(x - mn);
    m = mn;
  }
  __shared__ float sm[4][64], ss[4][64];
  sm[fg][nl] = m; ss[fg][nl] = s;
  __syncthreads();
  if (threadIdx.x < 64) {
    float M = sm[0][threadIdx.x], S = ss[0][threadIdx.x];
#pragma unroll
    for (int g = 1; g < 4; ++g) {
      const float mg = sm[g][threadIdx.x];
      const float Mn = fmaxf(M, mg);
      S = S * expf(M - Mn) + ss[g][threadIdx.x] * expf(mg - Mn);
      M = Mn;
    }
    cmax[h * cN + blockIdx.x * 64 + threadIdx.x] = M;
    csum[h * cN + blockIdx.x * 64 + threadIdx.x] = S;
  }
}

// ---------------------------------------------------------------------------
// p = softmax_F(logits) (bf16) and L1 rowsum over N per (h,f). grid (F, H)
// ---------------------------------------------------------------------------
__global__ __launch_bounds__(256) void pnorm_kernel(const float* __restrict__ logits,
                                                    const float* __restrict__ cmax,
                                                    const float* __restrict__ csum,
                                                    bf16* __restrict__ p,
                                                    float* __restrict__ rowsum) {
  const int f = blockIdx.x, h = blockIdx.y;
  const float* lrow = logits + ((long long)h * cF + f) * cN;
  bf16* prow = p + ((long long)h * cF + f) * cN;
  float acc = 0.0f;
  for (int n = threadIdx.x; n < cN; n += 256) {
    const float sv = expf(lrow[n] - cmax[h * cN + n]) / csum[h * cN + n];
    prow[n] = (bf16)sv;
    acc += sv;
  }
  __shared__ float red[256];
  red[threadIdx.x] = acc;
  __syncthreads();
  for (int s = 128; s > 0; s >>= 1) {
    if (threadIdx.x < s) red[threadIdx.x] += red[threadIdx.x + s];
    __syncthreads();
  }
  if (threadIdx.x == 0) rowsum[(long long)h * cF + f] = red[0];
}

// ---------------------------------------------------------------------------
// mean_attn partials: fixed-order column reduction of p[r,n]/(1e-7+rowsum[r])
// ---------------------------------------------------------------------------
__global__ __launch_bounds__(256) void colreduce_kernel(const bf16* __restrict__ p,
                                                        const float* __restrict__ rowsum,
                                                        float* __restrict__ part,
                                                        int rows_per_grp) {
  const int n  = blockIdx.x * 256 + threadIdx.x;
  const int r0 = blockIdx.y * rows_per_grp;
  float acc = 0.0f;
  for (int r = r0; r < r0 + rows_per_grp; ++r) {
    const float inv = 1.0f / (1e-7f + rowsum[r]);
    acc += (float)p[(long long)r * cN + n] * inv;
  }
  part[(long long)blockIdx.y * cN + n] = acc;
}

__global__ void meanfinal_kernel(const float* __restrict__ part, float* __restrict__ out,
                                 int ngroups, float scale) {
  const int n = blockIdx.x * 256 + threadIdx.x;
  if (n < cN) {
    float s = 0.0f;
    for (int g = 0; g < ngroups; ++g) s += part[(long long)g * cN + n];
    out[n] = s * scale;
  }
}

// ---------------------------------------------------------------------------
// Host orchestration
// ---------------------------------------------------------------------------
extern "C" void kernel_launch(void* const* d_in, const int* in_sizes, int n_in,
                              void* d_out, int out_size, void* d_ws, size_t ws_size,
                              hipStream_t stream) {
  const float* x       = (const float*)d_in[0];
  const float* seed    = (const float*)d_in[1];
  const float* norm1_w = (const float*)d_in[2];
  const float* norm1_b = (const float*)d_in[3];
  const float* nseed_w = (const float*)d_in[4];
  const float* nseed_b = (const float*)d_in[5];
  const float* trans_w = (const float*)d_in[6];
  const float* trans_b = (const float*)d_in[7];
  const float* w0_w    = (const float*)d_in[8];
  const float* w0_b    = (const float*)d_in[9];
  const float* w1_w    = (const float*)d_in[10];
  const float* w1_b    = (const float*)d_in[11];
  const float* proj_w  = (const float*)d_in[12];
  const float* proj_b  = (const float*)d_in[13];
  const float* norm2_w = (const float*)d_in[14];
  const float* norm2_b = (const float*)d_in[15];
  const float* fc1_w   = (const float*)d_in[16];
  const float* fc1_b   = (const float*)d_in[17];
  const float* fc2_w   = (const float*)d_in[18];
  const float* fc2_b   = (const float*)d_in[19];

  const long long BFC = (long long)cB * cF * cC;
  char* ws = (char*)d_ws;
  size_t off = 0;
  auto take = [&](size_t bytes) -> char* {
    off = (off + 255) & ~(size_t)255;
    char* ptr = ws + off;
    off += bytes;
    return ptr;
  };

  bf16*  xn      = (bf16*)take(BFC * 2);
  bf16*  qb      = (bf16*)take(BFC * 2);
  bf16*  snb     = (bf16*)take((size_t)cN * cC * 2);
  bf16*  w0b     = (bf16*)take((size_t)cN * cC * 2);
  bf16*  w1b     = (bf16*)take((size_t)cN * cC * 2);
  bf16*  w1t     = (bf16*)take((size_t)cC * cN * 2);   // w1b transposed [C,N]
  bf16*  wtT     = (bf16*)take((size_t)cC * cC * 2);   // weights stored transposed [N,K]
  bf16*  wt0     = (bf16*)take((size_t)cC * cC * 2);
  bf16*  wt1     = (bf16*)take((size_t)cC * cC * 2);
  bf16*  wtP     = (bf16*)take((size_t)cC * cC * 2);
  bf16*  wtF1    = (bf16*)take((size_t)cHID * cC * 2); // [HID, C]
  bf16*  wtF2    = (bf16*)take((size_t)cC * cHID * 2); // [C, HID]
  float* logits  = (float*)take((size_t)cH * cF * cN * 4);   // reused per b
  bf16*  pbuf    = (bf16*)take((size_t)cH * cF * cN * 2);    // reused per b
  float* cmax    = (float*)take((size_t)cH * cN * 4);
  float* csum    = (float*)take((size_t)cH * cN * 4);
  float* rsum    = (float*)take((size_t)cH * cF * 4);
  bf16*  attnout = (bf16*)take(BFC * 2);
  float* xres    = (float*)take(BFC * 4);
  bf16*  xn2     = (bf16*)take(BFC * 2);
  bf16*  hbuf    = (bf16*)take((size_t)cB * cF * cHID * 2);
  float* part    = (float*)take((size_t)cB * 32 * cN * 4);

  float* out_x    = (float*)d_out;
  float* out_mean = (float*)d_out + BFC;

  // 1) LayerNorms + transposed bf16 weight conversions
  ln512_kernel<<<cB * cF, 256, 0, stream>>>(x, norm1_w, norm1_b, xn);
  ln512_kernel<<<cN, 256, 0, stream>>>(seed, nseed_w, nseed_b, snb);
  cvt_bf16_t_kernel<<<dim3(cC / 32, cC / 32), 256, 0, stream>>>(trans_w, wtT, cC, cC);
  cvt_bf16_t_kernel<<<dim3(cC / 32, cC / 32), 256, 0, stream>>>(w0_w, wt0, cC, cC);
  cvt_bf16_t_kernel<<<dim3(cC / 32, cC / 32), 256, 0, stream>>>(w1_w, wt1, cC, cC);
  cvt_bf16_t_kernel<<<dim3(cC / 32, cC / 32), 256, 0, stream>>>(proj_w, wtP, cC, cC);
  cvt_bf16_t_kernel<<<dim3(cHID / 32, cC / 32), 256, 0, stream>>>(fc1_w, wtF1, cC, cHID);
  cvt_bf16_t_kernel<<<dim3(cC / 32, cHID / 32), 256, 0, stream>>>(fc2_w, wtF2, cHID, cC);

  GemmP g = {};

  // 2) q = xn @ trans_w + b  -> bf16
  g = {}; g.A = xn; g.lda = cC; g.B = wtT; g.ldb = cC; g.bias = trans_b;
  g.Ob = qb; g.ldc = cC; g.M = cB * cF; g.N = cC; g.K = cC;
  wmma_gemm_kernel<64, 128, 1><<<dim3(cC / 128, (cB * cF) / 64, 1), 256, 0, stream>>>(g);

  // 3) w0 = sn @ w0_w + b -> bf16 ; w1 likewise, then transpose w1 -> [C,N]
  g = {}; g.A = snb; g.lda = cC; g.B = wt0; g.ldb = cC; g.bias = w0_b;
  g.Ob = w0b; g.ldc = cC; g.M = cN; g.N = cC; g.K = cC;
  wmma_gemm_kernel<64, 128, 1><<<dim3(cC / 128, cN / 64, 1), 256, 0, stream>>>(g);
  g.B = wt1; g.bias = w1_b; g.Ob = w1b;
  wmma_gemm_kernel<64, 128, 1><<<dim3(cC / 128, cN / 64, 1), 256, 0, stream>>>(g);
  transpose_bf16_kernel<<<dim3(cC / 32, cN / 32), 256, 0, stream>>>(w1b, w1t, cN, cC);

  // 4) attention, one batch b at a time
  for (int b = 0; b < cB; ++b) {
    bf16* qb_b   = qb + (long long)b * cF * cC;
    bf16* aout_b = attnout + (long long)b * cF * cC;

    // logits[h,f,n] = sum_d q[b,f,h*64+d] * w0[n,h*64+d]  (B = w0b rows n)
    g = {}; g.A = qb_b; g.lda = cC; g.sA = cD; g.B = w0b; g.ldb = cC; g.sB = cD;
    g.Of = logits; g.ldc = cN; g.sC = (long long)cF * cN;
    g.M = cF; g.N = cN; g.K = cD;
    wmma_gemm_kernel<64, 128, 0><<<dim3(cN / 128, cF / 64, cH), 256, 0, stream>>>(g);

    colstats_kernel<<<dim3(cN / 64, cH), 256, 0, stream>>>(logits, cmax, csum);
    pnorm_kernel<<<dim3(cF, cH), 256, 0, stream>>>(logits, cmax, csum, pbuf, rsum);
    colreduce_kernel<<<dim3(cN / 256, 32), 256, 0, stream>>>(
        pbuf, rsum, part + (long long)b * 32 * cN, (cH * cF) / 32);

    // out[b,f,h*64+d] = (1/(1e-7+rowsum)) * sum_n p[h,f,n] * w1t[h*64+d, n]
    g = {}; g.A = pbuf; g.lda = cN; g.sA = (long long)cF * cN;
    g.B = w1t; g.ldb = cN; g.sB = (long long)cD * cN;
    g.Ob = aout_b; g.ldc = cC; g.sC = cD;
    g.rowsum = rsum; g.sRS = cF;
    g.M = cF; g.N = cD; g.K = cN;
    wmma_gemm_kernel<128, 64, 4><<<dim3(cD / 64, cF / 128, cH), 256, 0, stream>>>(g);
  }

  // 5) xres = x + attnout @ proj_w + proj_b
  g = {}; g.A = attnout; g.lda = cC; g.B = wtP; g.ldb = cC; g.bias = proj_b;
  g.Of = xres; g.ldc = cC; g.resid = x; g.ldr = cC;
  g.M = cB * cF; g.N = cC; g.K = cC;
  wmma_gemm_kernel<64, 128, 2><<<dim3(cC / 128, (cB * cF) / 64, 1), 256, 0, stream>>>(g);

  // 6) LN2, fc1 (+GELU), fc2 (+residual -> d_out)
  ln512_kernel<<<cB * cF, 256, 0, stream>>>(xres, norm2_w, norm2_b, xn2);

  g = {}; g.A = xn2; g.lda = cC; g.B = wtF1; g.ldb = cC; g.bias = fc1_b;
  g.Ob = hbuf; g.ldc = cHID; g.M = cB * cF; g.N = cHID; g.K = cC;
  wmma_gemm_kernel<64, 128, 3><<<dim3(cHID / 128, (cB * cF) / 64, 1), 256, 0, stream>>>(g);

  g = {}; g.A = hbuf; g.lda = cHID; g.B = wtF2; g.ldb = cHID; g.bias = fc2_b;
  g.Of = out_x; g.ldc = cC; g.resid = xres; g.ldr = cC;
  g.M = cB * cF; g.N = cC; g.K = cHID;
  wmma_gemm_kernel<64, 128, 2><<<dim3(cC / 128, (cB * cF) / 64, 1), 256, 0, stream>>>(g);

  // 7) mean_attn = sum(partials) / (B*H*F)
  meanfinal_kernel<<<dim3(cN / 256), 256, 0, stream>>>(
      part, out_mean, cB * 32, 1.0f / (float)(cB * cH * cF));
}
